// Attention_28845000359959
// MI455X (gfx1250) — matrix operand order repaired and checked by reading, not compiled, not run on hardware
//
#include <hip/hip_runtime.h>

// ---------------------------------------------------------------------------
// MI455X (gfx1250) multi-head attention, bf16 WMMA + async-LDS (wave32).
// cvt(x) -> transpose-cvt(weights) -> QKV GEMM (64x64/wave, scatter Q,K,V^T)
// -> cooperative flash attention (async double-buffered K/V panels in LDS)
// -> out GEMM (64x64/wave) + bias.
// ---------------------------------------------------------------------------

typedef __attribute__((ext_vector_type(16))) __bf16 v16bf;
typedef __attribute__((ext_vector_type(8)))  __bf16 v8bf;
typedef __attribute__((ext_vector_type(4)))  __bf16 v4bf;
typedef __attribute__((ext_vector_type(8)))  float  v8f;

#define BATCH   4
#define SEQ     2048
#define DIM     1024
#define HEADS   16
#define HDIM    64
#define ROWS    (BATCH * SEQ)      /* 8192 */
#define QKVCOLS (3 * DIM)          /* 3072 */
#define SCALE_Q 0.03125f           /* DIM^-0.5 = 1/32, folded into Q */

static __device__ __forceinline__ __bf16 f2bf(float f) {
    unsigned int u = __builtin_bit_cast(unsigned int, f);
    unsigned int r = (u + 0x7FFFu + ((u >> 16) & 1u)) >> 16;
    unsigned short s = (unsigned short)r;
    return __builtin_bit_cast(__bf16, s);
}

static __device__ __forceinline__ v16bf combine8(v8bf lo, v8bf hi) {
    return __builtin_shufflevector(lo, hi, 0,1,2,3,4,5,6,7,8,9,10,11,12,13,14,15);
}

static __device__ __forceinline__ v8f bfwmma(v16bf a, v16bf b, v8f c) {
    return __builtin_amdgcn_wmma_f32_16x16x32_bf16(false, a, false, b,
                                                   (short)0, c, false, false);
}

// gfx1250 async copy: 16B global -> LDS per lane, tracked by ASYNCcnt.
// LDS address = low 32 bits of the generic pointer (aperture rules, ISA 10.2).
static __device__ __forceinline__ void async_b128(void* lds, const void* gp) {
    unsigned l = (unsigned)(unsigned long long)lds;
    unsigned long long g = (unsigned long long)gp;
    asm volatile("global_load_async_to_lds_b128 %0, %1, off"
                 :: "v"(l), "v"(g) : "memory");
}
static __device__ __forceinline__ void wait_async0() {
    asm volatile("s_wait_asynccnt 0x0" ::: "memory");
}

// --------------------------- f32 -> bf16 convert ---------------------------
__global__ __launch_bounds__(256) void cvt_x_kernel(const float* __restrict__ in,
                                                    __bf16* __restrict__ out) {
    unsigned idx = (blockIdx.x * 256u + threadIdx.x) * 4u;
    float4 v = *(const float4*)(in + idx);
    v4bf o = { f2bf(v.x), f2bf(v.y), f2bf(v.z), f2bf(v.w) };
    *(v4bf*)(out + idx) = o;
}

// ----------------- f32 [rows][cols] -> bf16 [cols][rows] -------------------
__global__ __launch_bounds__(256) void transpose_cvt_kernel(const float* __restrict__ in,
                                                            __bf16* __restrict__ out,
                                                            unsigned rows, unsigned cols) {
    __shared__ float tile[32][33];
    unsigned tx = threadIdx.x, ty = threadIdx.y;
    unsigned icol = blockIdx.x * 32u + tx;
    #pragma unroll
    for (unsigned i = ty; i < 32u; i += 8u)
        tile[i][tx] = in[(size_t)(blockIdx.y * 32u + i) * cols + icol];
    __syncthreads();
    unsigned ocol = blockIdx.y * 32u + tx;   // == input row index
    #pragma unroll
    for (unsigned i = ty; i < 32u; i += 8u)
        out[(size_t)(blockIdx.x * 32u + i) * rows + ocol] = f2bf(tile[tx][i]);
}

// ------------------------------ QKV GEMM -----------------------------------
// 64x64 macro-tile per wave (4x4 WMMA tiles). Wt is K-contiguous: Wt[c][k].
// Each 64-col span == one head of one of {q,k,v}: which = nt>>4, h = nt&15.
__global__ __launch_bounds__(256) void qkv_gemm_kernel(const __bf16* __restrict__ X,
                                                       const __bf16* __restrict__ Wt,
                                                       __bf16* __restrict__ Q,
                                                       __bf16* __restrict__ K,
                                                       __bf16* __restrict__ Vt) {
    const unsigned lane = threadIdx.x & 31u, wave = threadIdx.x >> 5;
    const unsigned tile = blockIdx.x * 8u + wave;
    const unsigned NT   = QKVCOLS / 64u;                 // 48
    const unsigned mt = tile / NT, nt = tile % NT;
    const unsigned ln = lane & 15u, hi = lane >> 4;
    const unsigned kb8 = hi * 8u, kb16 = hi * 16u;

    const __bf16* ar[4]; const __bf16* br[4];
    #pragma unroll
    for (unsigned i = 0; i < 4; ++i) ar[i] = X  + (size_t)(mt * 64u + i * 16u + ln) * DIM;
    #pragma unroll
    for (unsigned j = 0; j < 4; ++j) br[j] = Wt + (size_t)(nt * 64u + j * 16u + ln) * DIM;

    v8f c[4][4] = {};
    for (unsigned k0 = 0; k0 < DIM; k0 += 32u) {
        v16bf a[4], b[4];
        #pragma unroll
        for (unsigned i = 0; i < 4; ++i)
            a[i] = *(const v16bf*)(ar[i] + k0 + kb16);
        #pragma unroll
        for (unsigned j = 0; j < 4; ++j)
            b[j] = *(const v16bf*)(br[j] + k0 + kb16);
        #pragma unroll
        for (unsigned i = 0; i < 4; ++i)
            #pragma unroll
            for (unsigned j = 0; j < 4; ++j)
                c[i][j] = bfwmma(a[i], b[j], c[i][j]);
    }

    const unsigned which = nt >> 4;                      // 0=q 1=k 2=v (uniform)
    const unsigned h  = nt & 15u;
    const unsigned b0 = (mt * 64u) / SEQ;                // uniform per wave
    const unsigned n0 = (mt * 64u) % SEQ;
    const size_t bh = (size_t)b0 * HEADS + h;
    #pragma unroll
    for (unsigned i = 0; i < 4; ++i) {
        #pragma unroll
        for (unsigned j = 0; j < 4; ++j) {
            const unsigned d = j * 16u + ln;
            #pragma unroll
            for (unsigned v = 0; v < 8; ++v) {
                const unsigned n = n0 + i * 16u + v + hi * 8u;
                float val = c[i][j][v];
                if (which == 0)
                    Q[(bh * SEQ + n) * HDIM + d] = f2bf(val * SCALE_Q);
                else if (which == 1)
                    K[(bh * SEQ + n) * HDIM + d] = f2bf(val);
                else
                    Vt[(bh * HDIM + d) * SEQ + n] = f2bf(val);
            }
        }
    }
}

// --------------------------- Flash attention -------------------------------
// Block = 8 waves sharing one (b,h); each wave owns a 16-query tile (128 q
// per block). 32-key K/V panels are async-copied into double-buffered LDS and
// consumed by all waves. S^T = K_blk @ Q^T (M=key, N=query) so softmax stats
// are per-lane (query = lane&15; lanes L and L^16 split the 16 key rows).
// O^T = V^T @ P^T keeps the per-query rescale per-lane.
__global__ __launch_bounds__(256) void attn_kernel(const __bf16* __restrict__ Q,
                                                   const __bf16* __restrict__ K,
                                                   const __bf16* __restrict__ Vt,
                                                   __bf16* __restrict__ AO) {
    __shared__ __bf16 lK[2][32][64];   // [buf][key][d]
    __shared__ __bf16 lV[2][64][32];   // [buf][d][key]

    const unsigned tid  = threadIdx.x;
    const unsigned lane = tid & 31u, wave = tid >> 5;
    const unsigned qsb  = blockIdx.x & 15u;              // 16 query superblocks
    const unsigned bh   = blockIdx.x >> 4;               // 64 (b,h) pairs
    const unsigned b    = bh >> 4, h = bh & 15u;

    const __bf16* Qp = Q  + (size_t)bh * SEQ * HDIM;
    const __bf16* Kp = K  + (size_t)bh * SEQ * HDIM;
    const __bf16* Vp = Vt + (size_t)bh * HDIM * SEQ;

    const unsigned ln = lane & 15u, hi = lane >> 4;
    const unsigned kb8 = hi * 8u, kb16 = hi * 16u;

    // Q^T B-fragments for d-windows [0,32) and [32,64)
    const unsigned qrow = qsb * 128u + wave * 16u + ln;
    v16bf bq0 = *(const v16bf*)(Qp + (size_t)qrow * HDIM + kb16);
    v16bf bq1 = *(const v16bf*)(Qp + (size_t)qrow * HDIM + 32u + kb16);

    // cooperative fill mapping: 256 threads x 16B == one 4KB panel each
    const unsigned kr = tid >> 3, kc = (tid & 7u) * 8u;  // K: 32 x 64
    const unsigned vr = tid >> 2, vc = (tid & 3u) * 8u;  // V: 64 x 32

    async_b128(&lK[0][kr][kc], Kp + (size_t)kr * HDIM + kc);
    async_b128(&lV[0][vr][vc], Vp + (size_t)vr * SEQ + vc);

    float m_run = -3.0e38f, l_run = 0.0f;
    v8f o[4] = {v8f{}, v8f{}, v8f{}, v8f{}};

    for (unsigned it = 0; it < SEQ / 32u; ++it) {
        const unsigned cur = it & 1u;
        wait_async0();
        __syncthreads();
        if (it + 1u < SEQ / 32u) {
            const unsigned nb = cur ^ 1u, njb = (it + 1u) * 32u;
            async_b128(&lK[nb][kr][kc], Kp + (size_t)(njb + kr) * HDIM + kc);
            async_b128(&lV[nb][vr][vc], Vp + (size_t)vr * SEQ + njb + vc);
        }

        // ---- scores: two 16-key tiles from LDS, d accumulated in 2 steps
        const __bf16* k0p = &lK[cur][ln][0];
        const __bf16* k1p = &lK[cur][16u + ln][0];
        v16bf a00 = *(const v16bf*)(k0p + kb16);
        v16bf a01 = *(const v16bf*)(k0p + 32u + kb16);
        v16bf a10 = *(const v16bf*)(k1p + kb16);
        v16bf a11 = *(const v16bf*)(k1p + 32u + kb16);
        v8f s0 = {}, s1 = {};
        s0 = bfwmma(a00, bq0, s0);  s0 = bfwmma(a01, bq1, s0);
        s1 = bfwmma(a10, bq0, s1);  s1 = bfwmma(a11, bq1, s1);

        // ---- online softmax (per-lane query column)
        float bm = s0[0];
        #pragma unroll
        for (int v = 0; v < 8; ++v) { bm = fmaxf(bm, s0[v]); bm = fmaxf(bm, s1[v]); }
        bm = fmaxf(bm, __shfl_xor(bm, 16, 32));
        float m_new = fmaxf(m_run, bm);
        float alpha = __expf(m_run - m_new);

        float p0[8], p1[8], psum = 0.0f;
        #pragma unroll
        for (int v = 0; v < 8; ++v) {
            p0[v] = __expf(s0[v] - m_new);
            p1[v] = __expf(s1[v] - m_new);
            psum += p0[v] + p1[v];
        }
        psum += __shfl_xor(psum, 16, 32);
        l_run = l_run * alpha + psum;
        m_run = m_new;
        #pragma unroll
        for (int t = 0; t < 4; ++t)
            #pragma unroll
            for (int v = 0; v < 8; ++v) o[t][v] *= alpha;

        // ---- P^T as B-fragment [32 keys x 16 queries]
        v16bf bp;
        #pragma unroll
        for (int j = 0; j < 8; ++j) {
            float x0 = __shfl_xor(p0[j], 16, 32);
            float x1 = __shfl_xor(p1[j], 16, 32);
            bp[j]     = f2bf(hi ? x1 : p0[j]);
            bp[j + 8] = f2bf(hi ? p1[j] : x0);
        }

        // ---- O^T += V^T_blk @ P^T (4 d-tiles from LDS)
        #pragma unroll
        for (int t = 0; t < 4; ++t) {
            const __bf16* vrp = &lV[cur][t * 16 + ln][0];
            v16bf av = *(const v16bf*)(vrp + kb16);
            o[t] = bfwmma(av, bp, o[t]);
        }
    }

    const float invl = 1.0f / l_run;
    __bf16* outp = AO + ((size_t)b * SEQ + qsb * 128u + wave * 16u + ln) * DIM + h * HDIM;
    #pragma unroll
    for (int t = 0; t < 4; ++t)
        #pragma unroll
        for (int v = 0; v < 8; ++v)
            outp[t * 16 + v + hi * 8] = f2bf(o[t][v] * invl);
}

// ------------------------- Output projection -------------------------------
// 64x64 macro-tile per wave. Y f32 = AO bf16 @ Wout + bias.
__global__ __launch_bounds__(256) void out_gemm_kernel(const __bf16* __restrict__ A,
                                                       const __bf16* __restrict__ Wt,
                                                       const float* __restrict__ bias,
                                                       float* __restrict__ Y) {
    const unsigned lane = threadIdx.x & 31u, wave = threadIdx.x >> 5;
    const unsigned tile = blockIdx.x * 8u + wave;
    const unsigned NT   = DIM / 64u;                     // 16
    const unsigned mt = tile / NT, nt = tile % NT;
    const unsigned ln = lane & 15u, hi = lane >> 4;
    const unsigned kb8 = hi * 8u, kb16 = hi * 16u;

    const __bf16* ar[4]; const __bf16* br[4];
    #pragma unroll
    for (unsigned i = 0; i < 4; ++i) ar[i] = A  + (size_t)(mt * 64u + i * 16u + ln) * DIM;
    #pragma unroll
    for (unsigned j = 0; j < 4; ++j) br[j] = Wt + (size_t)(nt * 64u + j * 16u + ln) * DIM;

    v8f c[4][4] = {};
    for (unsigned k0 = 0; k0 < DIM; k0 += 32u) {
        v16bf a[4], b[4];
        #pragma unroll
        for (unsigned i = 0; i < 4; ++i)
            a[i] = *(const v16bf*)(ar[i] + k0 + kb16);
        #pragma unroll
        for (unsigned j = 0; j < 4; ++j)
            b[j] = *(const v16bf*)(br[j] + k0 + kb16);
        #pragma unroll
        for (unsigned i = 0; i < 4; ++i)
            #pragma unroll
            for (unsigned j = 0; j < 4; ++j)
                c[i][j] = bfwmma(a[i], b[j], c[i][j]);
    }

    #pragma unroll
    for (unsigned j = 0; j < 4; ++j) {
        const unsigned col = nt * 64u + j * 16u + ln;
        const float bv = bias[col];
        #pragma unroll
        for (unsigned i = 0; i < 4; ++i)
            #pragma unroll
            for (unsigned v = 0; v < 8; ++v)
                Y[(size_t)(mt * 64u + i * 16u + v + hi * 8u) * DIM + col] = c[i][j][v] + bv;
    }
}

// ---------------------------------------------------------------------------
extern "C" void kernel_launch(void* const* d_in, const int* in_sizes, int n_in,
                              void* d_out, int out_size, void* d_ws, size_t ws_size,
                              hipStream_t stream) {
    const float* x     = (const float*)d_in[0];   // [4,2048,1024]
    const float* w_qkv = (const float*)d_in[1];   // [1024,3072]
    const float* w_out = (const float*)d_in[2];   // [1024,1024]
    const float* b_out = (const float*)d_in[3];   // [1024]
    float* out = (float*)d_out;

    char* ws = (char*)d_ws;
    __bf16* Xbf   = (__bf16*)(ws + (size_t)0);           // 16 MB
    __bf16* Wqkvt = (__bf16*)(ws + ((size_t)16 << 20));  //  6 MB  [3072][1024]
    __bf16* Woutt = (__bf16*)(ws + ((size_t)22 << 20));  //  2 MB  [1024][1024]
    __bf16* Qb    = (__bf16*)(ws + ((size_t)24 << 20));  // 16 MB  [b][h][n][d]
    __bf16* Kb    = (__bf16*)(ws + ((size_t)40 << 20));  // 16 MB  [b][h][n][d]
    __bf16* Vtb   = (__bf16*)(ws + ((size_t)56 << 20));  // 16 MB  [b][h][d][n]
    __bf16* AO    = (__bf16*)(ws + ((size_t)72 << 20));  // 16 MB  [b][n][dim]

    cvt_x_kernel<<<ROWS * DIM / (256 * 4), 256, 0, stream>>>(x, Xbf);

    dim3 tb(32, 8);
    transpose_cvt_kernel<<<dim3(QKVCOLS / 32, DIM / 32), tb, 0, stream>>>(w_qkv, Wqkvt, DIM, QKVCOLS);
    transpose_cvt_kernel<<<dim3(DIM / 32, DIM / 32), tb, 0, stream>>>(w_out, Woutt, DIM, DIM);

    qkv_gemm_kernel<<<(ROWS / 64) * (QKVCOLS / 64) / 8, 256, 0, stream>>>(Xbf, Wqkvt, Qb, Kb, Vtb);

    attn_kernel<<<BATCH * HEADS * (SEQ / 128), 256, 0, stream>>>(Qb, Kb, Vtb, AO);

    out_gemm_kernel<<<(ROWS / 64) * (DIM / 64) / 8, 256, 0, stream>>>(AO, Woutt, b_out, out);
}
